// SentenceEncoder_15393162789368
// MI455X (gfx1250) — matrix operand order repaired and checked by reading, not compiled, not run on hardware
//
#include <hip/hip_runtime.h>
#include <hip/hip_bf16.h>
#include <math.h>

// ---------------- model dims ----------------
#define WDIM   300
#define HDIM   100
#define BATCH  128
#define SLEN   48
#define CLEN   16
#define CDIM   16
#define GDIM   400      // 4*H gates
#define XLD    1120     // padded max concat feature dim (1116 -> 1120)
#define FCK    512      // padded feature K for FC (500 -> 512)
#define FCN    800

// GEMM block tiling: 8 waves, wave w owns M-tile w and all 5 N-tiles (guard-free)
#define BM     128
#define BN     80
#define KPANEL 32

typedef __attribute__((ext_vector_type(16))) _Float16 v16h;
typedef __attribute__((ext_vector_type(8)))  _Float16 v8h;
typedef __attribute__((ext_vector_type(8)))  float    v8f;
typedef unsigned int u32x4 __attribute__((ext_vector_type(4)));
typedef int          i32x4 __attribute__((ext_vector_type(4)));
typedef int          i32x8 __attribute__((ext_vector_type(8)));

union HFrag { v16h v; v8h h[2]; };

__device__ inline v16h load_a_frag(const _Float16* rowptr, int lane) {
    int kb = (lane < 16) ? 0 : 8;
    HFrag u;
    u.h[0] = *(const v8h*)(rowptr + kb);
    u.h[1] = *(const v8h*)(rowptr + kb + 16);
    return u.v;
}
__device__ inline v16h load_b_frag(const _Float16* rowptr, int lane) {
    int kb = (lane < 16) ? 0 : 8;
    HFrag u;
    u.h[0] = *(const v8h*)(rowptr + kb);
    u.h[1] = *(const v8h*)(rowptr + kb + 16);
    return u.v;
}

__device__ inline float sigmoidf_(float x) { return 1.0f / (1.0f + __expf(-x)); }

// ---------------- TDM: 2D tile load Global -> LDS ----------------
__device__ inline void tdm_load_2d(_Float16* ldsptr, const _Float16* gptr,
                                   int rows, int stride_elems) {
#if __has_builtin(__builtin_amdgcn_tensor_load_to_lds)
    unsigned long long ga = (unsigned long long)(size_t)gptr;
    u32x4 g0;
    g0[0] = 1u;                                              // count=1, flags 0
    g0[1] = (unsigned int)(size_t)ldsptr;                    // LDS byte address
    g0[2] = (unsigned int)ga;                                // global_addr[31:0]
    g0[3] = (unsigned int)((ga >> 32) & 0x01FFFFFFu) | (2u << 30);  // addr[56:32] | type=2
    i32x8 g1;
    g1[0] = 0x00010000;                 // workgroup_mask=0, data_size=1 (2 bytes)
    g1[1] = 0;                          // no atomic barrier; tensor_dim0[15:0]=0
    g1[2] = 0x4000;                     // tensor_dim0=0x40000000 (no OOB)
    g1[3] = (KPANEL << 16) | 0x4000;    // tensor_dim1=0x40000000, tile_dim0=32
    g1[4] = rows;                       // tile_dim1=rows, tile_dim2=0
    g1[5] = stride_elems;               // tensor_dim0_stride
    g1[6] = 0;
    g1[7] = 0;
    i32x4 z4 = {};
#if defined(__clang_major__) && (__clang_major__ >= 23)
    i32x8 z8 = {};
    __builtin_amdgcn_tensor_load_to_lds(g0, g1, z4, z4, z8, 0);
#else
    __builtin_amdgcn_tensor_load_to_lds(g0, g1, z4, z4, 0);
#endif
#else
    for (int i = (int)(threadIdx.x & 31); i < rows * KPANEL; i += 32)
        ldsptr[i] = gptr[(size_t)(i / KPANEL) * stride_elems + (i % KPANEL)];
#endif
}

__device__ inline void tdm_wait_2() {
#if __has_builtin(__builtin_amdgcn_s_wait_tensorcnt)
    __builtin_amdgcn_s_wait_tensorcnt(2);
#endif
}
__device__ inline void tdm_wait_0() {
#if __has_builtin(__builtin_amdgcn_s_wait_tensorcnt)
    __builtin_amdgcn_s_wait_tensorcnt(0);
#endif
}

// ---------------- utility: zero fill ----------------
__global__ void fill_zero_kernel(unsigned int* p, long n) {
    long i = (long)blockIdx.x * blockDim.x + threadIdx.x;
    if (i < n) p[i] = 0u;
}

// ---------------- weight convert + pad to f16 ----------------
__global__ void convert_pad_kernel(const float* __restrict__ src, _Float16* __restrict__ dst,
                                   int N, int K, int Kpad) {
    int idx = blockIdx.x * blockDim.x + threadIdx.x;
    if (idx >= N * Kpad) return;
    int n = idx / Kpad, k = idx % Kpad;
    dst[idx] = (k < K) ? (_Float16)src[n * K + k] : (_Float16)0.0f;
}

__global__ void bias_sum_kernel(const float* a, const float* b, float* o, int n) {
    int i = blockIdx.x * blockDim.x + threadIdx.x;
    if (i < n) o[i] = a[i] + b[i];
}

// ---------------- embedding: word gather + char conv/maxpool ----------------
__global__ void embed_kernel(const int* __restrict__ words, const int* __restrict__ chars,
                             const float* __restrict__ wemb, const float* __restrict__ cemb,
                             const float* __restrict__ convw, const float* __restrict__ convb,
                             _Float16* __restrict__ xdst) {
    __shared__ float ce[CLEN * CDIM];
    __shared__ float red[64];
    int b = blockIdx.x / SLEN, s = blockIdx.x % SLEN;
    int t = threadIdx.x;
    for (int idx = t; idx < CLEN * CDIM; idx += 64) {
        int c = idx / CDIM, d = idx % CDIM;
        int id = chars[(b * SLEN + s) * CLEN + c];
        ce[idx] = cemb[id * CDIM + d];
    }
    __syncthreads();
    _Float16* row = xdst + ((size_t)s * BATCH + b) * XLD;
    int wid = words[b * SLEN + s];
    for (int col = t; col < WDIM; col += 64)
        row[col] = (_Float16)wemb[(size_t)wid * WDIM + col];
    int oc = t >> 2, g = t & 3;
    float w0 = convw[oc * 3 + 0], w1 = convw[oc * 3 + 1], w2 = convw[oc * 3 + 2];
    float bb = convb[oc];
    float mx = -1e30f;
    for (int p = g; p < CLEN * CDIM - 2; p += 4) {
        float v = bb + w0 * ce[p] + w1 * ce[p + 1] + w2 * ce[p + 2];
        if (b != 0) v = fmaxf(v, 0.0f);      // reference quirk: batch row 0 skips relu
        mx = fmaxf(mx, v);
    }
    red[t] = mx;
    __syncthreads();
    if (g == 0) {
        float m = fmaxf(fmaxf(red[t], red[t + 1]), fmaxf(red[t + 2], red[t + 3]));
        row[WDIM + oc] = (_Float16)m;
    }
}

// ---------------- TDM-staged WMMA GEMM: out = A[M,K] * W[N,K]^T + bias ------------
// 256 threads (8 waves); wave w computes M-tile w x all 5 N-tiles (no guards).
// A fragment loaded once per K-step, reused across 5 WMMAs.
__global__ void wmma_gemm_tdm_kernel(const _Float16* __restrict__ A, int lda,
                                     const _Float16* __restrict__ W, int ldw,
                                     const float* __restrict__ bias,
                                     float* __restrict__ outF, _Float16* __restrict__ outH,
                                     int ldo, int Nblocks, int Ksteps, int relu) {
    __shared__ _Float16 sA[2][BM * KPANEL];
    __shared__ _Float16 sW[2][BN * KPANEL];
    int tid = threadIdx.x;
    int wv = tid >> 5, lane = tid & 31;
    int mb = blockIdx.x / Nblocks, nb = blockIdx.x % Nblocks;
    int m0 = mb * BM, n0 = nb * BN;

    v8f acc[5] = {};

    if (wv == 0) {        // TDM ignores EXEC; issue from one wave only
        tdm_load_2d(&sA[0][0], A + (size_t)m0 * lda, BM, lda);
        tdm_load_2d(&sW[0][0], W + (size_t)n0 * ldw, BN, ldw);
    }
    for (int kt = 0; kt < Ksteps; ++kt) {
        int buf = kt & 1;
        if (wv == 0) {
            if (kt + 1 < Ksteps) {
                tdm_load_2d(&sA[buf ^ 1][0], A + (size_t)m0 * lda + (kt + 1) * KPANEL, BM, lda);
                tdm_load_2d(&sW[buf ^ 1][0], W + (size_t)n0 * ldw + (kt + 1) * KPANEL, BN, ldw);
                tdm_wait_2();     // drain panel kt, keep kt+1 in flight
            } else {
                tdm_wait_0();
            }
        }
        __syncthreads();
        v16h af = load_a_frag(&sA[buf][(wv * 16 + (lane & 15)) * KPANEL], lane);
#pragma unroll
        for (int t = 0; t < 5; ++t) {
            v16h bf = load_b_frag(&sW[buf][(t * 16 + (lane & 15)) * KPANEL], lane);
            acc[t] = __builtin_amdgcn_wmma_f32_16x16x32_f16(false, af, false, bf,
                                                            (short)0, acc[t], false, false);
        }
        __syncthreads();
    }
#pragma unroll
    for (int t = 0; t < 5; ++t) {
        int n = n0 + t * 16 + (lane & 15);
        float bn = bias ? bias[n] : 0.0f;
        int rbase = m0 + wv * 16 + ((lane >> 4) << 3);
#pragma unroll
        for (int r = 0; r < 8; ++r) {
            float v = acc[t][r] + bn;
            if (relu) v = fmaxf(v, 0.0f);
            if (outF) outF[(size_t)(rbase + r) * ldo + n] = v;
            else      outH[(size_t)(rbase + r) * ldo + n] = (_Float16)v;
        }
    }
}

// ---------------- persistent LSTM scan (one workgroup, Whh resident in LDS) --------
// wave w: fixed M-tile (w&7), N-tiles (w>>3)+4t; nt<25 guard scalarized via
// readfirstlane so it lowers to s_cmp/s_cbranch (no EXEC churn around WMMA).
__global__ void lstm_scan_kernel(const float* __restrict__ xproj,
                                 const float* __restrict__ whh,
                                 float* __restrict__ hs,
                                 _Float16* __restrict__ xcat, int coff) {
    extern __shared__ char smem[];
    _Float16* sW = (_Float16*)smem;          // [400][128] padded recurrent weights
    _Float16* sH = sW + GDIM * 128;          // [128][128] padded hidden state
    _Float16* sG = sH + 128 * 128;           // [128][400] gate pre-activations
    int tid = threadIdx.x;
    for (int idx = tid; idx < GDIM * 128; idx += 1024) {
        int nn = idx >> 7, kk = idx & 127;
        sW[idx] = (kk < HDIM) ? (_Float16)whh[nn * HDIM + kk] : (_Float16)0.0f;
    }
    for (int idx = tid; idx < 128 * 128; idx += 1024) sH[idx] = (_Float16)0.0f;
    float c[13];
#pragma unroll
    for (int j = 0; j < 13; ++j) c[j] = 0.0f;
    int wv = tid >> 5, lane = tid & 31;
    int mt = wv & 7;                          // M-tile (fixed per wave)
    int ntb = wv >> 3;                        // N-tile base 0..3
    int mrow = mt * 16 + (lane & 15);
    for (int s = 0; s < SLEN; ++s) {
        __syncthreads();
        v8f acc[7] = {};
#pragma unroll
        for (int kt = 0; kt < 4; ++kt) {
            v16h af = load_a_frag(sH + mrow * 128 + kt * 32, lane);
#pragma unroll
            for (int t = 0; t < 7; ++t) {
                int nt = ntb + 4 * t;
                if (__builtin_amdgcn_readfirstlane(nt) < 25) {
                    v16h bf = load_b_frag(sW + (nt * 16 + (lane & 15)) * 128 + kt * 32, lane);
                    acc[t] = __builtin_amdgcn_wmma_f32_16x16x32_f16(false, af, false, bf,
                                                                    (short)0, acc[t], false, false);
                }
            }
        }
        const float* xp = xproj + (size_t)s * BATCH * GDIM;
        int rbase = mt * 16 + ((lane >> 4) << 3);
#pragma unroll
        for (int t = 0; t < 7; ++t) {
            int nt = ntb + 4 * t;
            if (__builtin_amdgcn_readfirstlane(nt) < 25) {
                int n = nt * 16 + (lane & 15);
#pragma unroll
                for (int r = 0; r < 8; ++r) {
                    int row = rbase + r;
                    sG[row * GDIM + n] = (_Float16)(acc[t][r] + xp[row * GDIM + n]);
                }
            }
        }
        __syncthreads();
        for (int j = 0; j < 13; ++j) {
            int idx = tid + (j << 10);
            if (idx < BATCH * HDIM) {
                int b = idx / HDIM, hc = idx % HDIM;
                float gi = (float)sG[b * GDIM + hc];
                float gf = (float)sG[b * GDIM + 100 + hc];
                float gg = (float)sG[b * GDIM + 200 + hc];
                float go = (float)sG[b * GDIM + 300 + hc];
                float cv = sigmoidf_(gf) * c[j] + sigmoidf_(gi) * tanhf(gg);
                c[j] = cv;
                float h = sigmoidf_(go) * tanhf(cv);
                sH[b * 128 + hc] = (_Float16)h;
                hs[((size_t)s * BATCH + b) * HDIM + hc] = h;
                xcat[((size_t)s * BATCH + b) * XLD + coff + hc] = (_Float16)h;
            }
        }
    }
}

// ---------------- co-attention: a_p[b,p,:] = softmax_q(hp.hh) @ hh -------------
__global__ void attn_kernel(const float* __restrict__ hp, const float* __restrict__ hh,
                            _Float16* __restrict__ xdst, int coff) {
    __shared__ float sc[SLEN];
    __shared__ float srow[HDIM];
    int b = blockIdx.x / SLEN, p = blockIdx.x % SLEN;
    int t = threadIdx.x;
    if (t < HDIM) srow[t] = hp[((size_t)p * BATCH + b) * HDIM + t];
    __syncthreads();
    if (t < SLEN) {
        float s = 0.0f;
        for (int h = 0; h < HDIM; ++h) s += srow[h] * hh[((size_t)t * BATCH + b) * HDIM + h];
        sc[t] = s;
    }
    __syncthreads();
    float mx = -1e30f;
    for (int q = 0; q < SLEN; ++q) mx = fmaxf(mx, sc[q]);
    float sum = 0.0f;
    for (int q = 0; q < SLEN; ++q) sum += __expf(sc[q] - mx);
    if (t < HDIM) {
        float acc = 0.0f;
        for (int q = 0; q < SLEN; ++q)
            acc += __expf(sc[q] - mx) * hh[((size_t)q * BATCH + b) * HDIM + t];
        xdst[((size_t)p * BATCH + b) * XLD + coff + t] = (_Float16)(acc / sum);
    }
}

// ---------------- max-pool over time + feature construction -------------------
__global__ void pool_feats_kernel(const float* __restrict__ hsP, const float* __restrict__ hsH,
                                  _Float16* __restrict__ feats) {
    int b = blockIdx.x, t = threadIdx.x;
    if (t < 12) feats[b * FCK + 500 + t] = (_Float16)0.0f;
    if (t >= HDIM) return;
    float hq = -1e30f, hv = -1e30f;
    for (int s = 0; s < SLEN; ++s) {
        hq = fmaxf(hq, hsP[((size_t)s * BATCH + b) * HDIM + t]);
        hv = fmaxf(hv, hsH[((size_t)s * BATCH + b) * HDIM + t]);
    }
    _Float16* f = feats + (size_t)b * FCK;
    f[t]       = (_Float16)hq;
    f[100 + t] = (_Float16)hv;
    f[200 + t] = (_Float16)(hv - hq);
    f[300 + t] = (_Float16)(hq * hv);
    f[400 + t] = (_Float16)fabsf(hq - hv);
}

// ---------------- final 2-way classifier ----------------
__global__ void last_kernel(const _Float16* __restrict__ fc1, const float* __restrict__ lw,
                            const float* __restrict__ lb, float* __restrict__ out) {
    int t = threadIdx.x;
    int b = t >> 1, cls = t & 1;
    float acc = lb[cls];
    for (int j = 0; j < FCN; ++j) acc += (float)fc1[(size_t)b * FCN + j] * lw[cls * FCN + j];
    out[b * 2 + cls] = sigmoidf_(acc);
}

// ================================================================================
extern "C" void kernel_launch(void* const* d_in, const int* in_sizes, int n_in,
                              void* d_out, int out_size, void* d_ws, size_t ws_size,
                              hipStream_t stream) {
    (void)in_sizes; (void)n_in; (void)out_size; (void)ws_size;
    const int*   q_words = (const int*)d_in[0];
    const int*   q_chars = (const int*)d_in[1];
    const int*   s_words = (const int*)d_in[2];
    const int*   s_chars = (const int*)d_in[3];
    const float* wemb    = (const float*)d_in[4];
    const float* cemb    = (const float*)d_in[5];
    const float* convw   = (const float*)d_in[6];
    const float* convb   = (const float*)d_in[7];
    const float* fclw    = (const float*)d_in[28];
    const float* fclb    = (const float*)d_in[29];
    const float* lastw   = (const float*)d_in[30];
    const float* lastb   = (const float*)d_in[31];

    char* ws = (char*)d_ws;
    auto alloc = [&](size_t bytes) { char* p = ws; ws += (bytes + 255) & ~(size_t)255; return p; };
    const size_t XHALVES = (size_t)SLEN * BATCH * XLD;
    _Float16* px16    = (_Float16*)alloc(XHALVES * 2);
    _Float16* hx16    = (_Float16*)alloc(XHALVES * 2);
    _Float16* wih16   = (_Float16*)alloc((size_t)GDIM * XLD * 2);
    float*    gbias   = (float*)   alloc(GDIM * 4);
    float*    xprojP  = (float*)   alloc((size_t)SLEN * BATCH * GDIM * 4);
    float*    xprojH  = (float*)   alloc((size_t)SLEN * BATCH * GDIM * 4);
    float*    hsP     = (float*)   alloc((size_t)SLEN * BATCH * HDIM * 4);
    float*    hsH     = (float*)   alloc((size_t)SLEN * BATCH * HDIM * 4);
    _Float16* feats16 = (_Float16*)alloc((size_t)BATCH * FCK * 2);
    _Float16* fclw16  = (_Float16*)alloc((size_t)FCN * FCK * 2);
    _Float16* fc1_16  = (_Float16*)alloc((size_t)BATCH * FCN * 2);

    long zn = (long)(XHALVES / 2);
    fill_zero_kernel<<<(unsigned)((zn + 255) / 256), 256, 0, stream>>>((unsigned int*)px16, zn);
    fill_zero_kernel<<<(unsigned)((zn + 255) / 256), 256, 0, stream>>>((unsigned int*)hx16, zn);

    embed_kernel<<<BATCH * SLEN, 64, 0, stream>>>(q_words, q_chars, wemb, cemb, convw, convb, px16);
    embed_kernel<<<BATCH * SLEN, 64, 0, stream>>>(s_words, s_chars, wemb, cemb, convw, convb, hx16);

    const size_t SMEM_SCAN = (size_t)(GDIM * 128 + 128 * 128 + 128 * GDIM) * 2;  // 237,568 B

    for (int k = 0; k < 5; ++k) {
        const float* wih = (const float*)d_in[8 + 4 * k];
        const float* whh = (const float*)d_in[9 + 4 * k];
        const float* bih = (const float*)d_in[10 + 4 * k];
        const float* bhh = (const float*)d_in[11 + 4 * k];
        int D = WDIM + CDIM + 2 * HDIM * k;          // 316 + 200k
        int Ksteps = ((D + 31) & ~31) / 32;

        convert_pad_kernel<<<(GDIM * XLD + 255) / 256, 256, 0, stream>>>(wih, wih16, GDIM, D, XLD);
        bias_sum_kernel<<<2, 256, 0, stream>>>(bih, bhh, gbias, GDIM);

        int Mblocks = SLEN * BATCH / BM;   // 48
        int Nblocks = GDIM / BN;           // 5
        wmma_gemm_tdm_kernel<<<Mblocks * Nblocks, 256, 0, stream>>>(
            px16, XLD, wih16, XLD, gbias, xprojP, nullptr, GDIM, Nblocks, Ksteps, 0);
        wmma_gemm_tdm_kernel<<<Mblocks * Nblocks, 256, 0, stream>>>(
            hx16, XLD, wih16, XLD, gbias, xprojH, nullptr, GDIM, Nblocks, Ksteps, 0);

        lstm_scan_kernel<<<1, 1024, SMEM_SCAN, stream>>>(xprojP, whh, hsP, px16, D + HDIM);
        lstm_scan_kernel<<<1, 1024, SMEM_SCAN, stream>>>(xprojH, whh, hsH, hx16, D + HDIM);

        attn_kernel<<<BATCH * SLEN, 128, 0, stream>>>(hsP, hsH, px16, D);
        attn_kernel<<<BATCH * SLEN, 128, 0, stream>>>(hsH, hsP, hx16, D);
    }

    pool_feats_kernel<<<BATCH, 128, 0, stream>>>(hsP, hsH, feats16);
    convert_pad_kernel<<<(FCN * FCK + 255) / 256, 256, 0, stream>>>(fclw, fclw16, FCN, 500, FCK);
    wmma_gemm_tdm_kernel<<<(BATCH / BM) * (FCN / BN), 256, 0, stream>>>(
        feats16, FCK, fclw16, FCK, fclb, nullptr, fc1_16, FCN, FCN / BN, FCK / 32, 1);
    last_kernel<<<1, 256, 0, stream>>>(fc1_16, lastw, lastb, (float*)d_out);
}
